// SimpleGRU_68848325755025
// MI455X (gfx1250) — matrix operand order repaired and checked by reading, not compiled, not run on hardware
//
#include <hip/hip_runtime.h>
#include <hip/hip_bf16.h>
#include <stdint.h>

typedef __attribute__((ext_vector_type(16))) __bf16 v16bf;
typedef __attribute__((ext_vector_type(8)))  __bf16 v8bf;
typedef __attribute__((ext_vector_type(8)))  float  v8f;

#define GB 64
#define GS 2048
#define GI 256
#define GH 512
#define GO 10

// ---------- helpers ----------

__device__ __forceinline__ v8f wmma_bf16(v16bf a, v16bf b, v8f c) {
  return __builtin_amdgcn_wmma_f32_16x16x32_bf16(false, a, false, b, (short)0, c,
                                                 false, false);
}

// A-matrix (16x32 bf16) load from LDS row-major [16][stride] at K-offset k0.
// lane<16 : row=lane,  K = k0+0..7  and k0+16..23
// lane>=16: row=lane-16,K = k0+8..15 and k0+24..31
__device__ __forceinline__ v16bf load_A_lds(const __bf16* base, int lane, int k0,
                                            int stride) {
  int m = lane & 15;
  int s = (lane >> 4) * 8;
  const __bf16* p = base + m * stride + k0 + s;
  v8bf lo = *(const v8bf*)p;
  v8bf hi = *(const v8bf*)(p + 16);
  v16bf a;
#pragma unroll
  for (int i = 0; i < 8; ++i) { a[i] = lo[i]; a[i + 8] = hi[i]; }
  return a;
}

// A-matrix load from global f32 row-major [.][K], converting to bf16.
__device__ __forceinline__ v16bf load_A_f32(const float* X, int mBase, int k0,
                                            int K, int lane) {
  int m = mBase + (lane & 15);
  int s = (lane >> 4) * 8;
  const float* p = X + (size_t)m * K + k0 + s;
  v16bf a;
#pragma unroll
  for (int i = 0; i < 8; ++i) {
    a[i]     = (__bf16)p[i];
    a[i + 8] = (__bf16)p[i + 16];
  }
  return a;
}

// B-matrix (32x16) from row-major weight U[N][K] (B[k][n] = U[n][k]).
// lane<16 : col n = nBase+lane, K=k0..k0+7 and k0+16..23 ; lane>=16 : same col, K=k0+8..15 and k0+24..31
__device__ __forceinline__ v16bf load_B_w(const __bf16* U, int nBase, int k0,
                                          int K, int lane) {
  int n  = nBase + (lane & 15);
  int ks = k0 + (lane >> 4) * 8;
  const __bf16* p = U + (size_t)n * K + ks;
  v8bf lo = *(const v8bf*)p;
  v8bf hi = *(const v8bf*)(p + 16);
  v16bf b;
#pragma unroll
  for (int i = 0; i < 8; ++i) { b[i] = lo[i]; b[i + 8] = hi[i]; }
  return b;
}

__device__ __forceinline__ float fast_sigmoid(float x) {
  return 1.0f / (1.0f + __expf(-x));
}
__device__ __forceinline__ float fast_tanh(float x) {
  return 1.0f - 2.0f / (__expf(2.0f * x) + 1.0f);
}

// ---- CDNA5 async global->LDS copy (ASYNCcnt-tracked), via inline asm ----
// VDST VGPR holds the LDS byte address (low 32 bits of the flat shared addr),
// VADDR holds the 64-bit global address (GV mode).
__device__ __forceinline__ void async_b128(__bf16* lds_dst, const __bf16* gsrc) {
  uint32_t lds_off = (uint32_t)(uintptr_t)lds_dst;
  asm volatile("global_load_async_to_lds_b128 %0, %1, off"
               :
               : "v"(lds_off), "v"(gsrc)
               : "memory");
}
__device__ __forceinline__ void wait_async0() {
  asm volatile("s_wait_asynccnt 0x0" ::: "memory");
}

// ---------- kernels ----------

__global__ void gru_cvt_bf16(const float* __restrict__ src,
                             __bf16* __restrict__ dst, int n) {
  int i = blockIdx.x * blockDim.x + threadIdx.x;
  if (i < n) dst[i] = (__bf16)src[i];
}

// Input projection: out[g] = x @ W[g]^T + b[g], stored bf16 [B*S][H].
// grid.x enumerates tiles (8 waves/block, 1 16x16 tile per wave), grid.y = gate.
__global__ __launch_bounds__(256) void gru_proj(
    const float* __restrict__ x,
    const __bf16* __restrict__ Wzb, const __bf16* __restrict__ Wrb,
    const __bf16* __restrict__ Whb,
    const float* __restrict__ bz, const float* __restrict__ br,
    const float* __restrict__ bh,
    __bf16* __restrict__ xzb, __bf16* __restrict__ xrb,
    __bf16* __restrict__ xhb) {
  const int lane = threadIdx.x & 31;
  const int wave = threadIdx.x >> 5;
  const int g = blockIdx.y;
  const __bf16* Wb = (g == 0) ? Wzb : ((g == 1) ? Wrb : Whb);
  const float*  bv = (g == 0) ? bz  : ((g == 1) ? br  : bh);
  __bf16*      out = (g == 0) ? xzb : ((g == 1) ? xrb : xhb);

  const int tile  = blockIdx.x * 8 + wave;   // 0 .. 8192*32-1
  const int tileN = tile & 31;               // H/16 = 32 tiles
  const int tileM = tile >> 5;               // (B*S)/16 = 8192 tiles
  const int mBase = tileM * 16;
  const int nBase = tileN * 16;
  const int col   = nBase + (lane & 15);

  const float bval = bv[col];
  v8f acc;
#pragma unroll
  for (int v = 0; v < 8; ++v) acc[v] = bval;

#pragma unroll
  for (int k0 = 0; k0 < GI; k0 += 32) {
    v16bf a = load_A_f32(x, mBase, k0, GI, lane);
    v16bf b = load_B_w(Wb, nBase, k0, GI, lane);
    acc = wmma_bf16(a, b, acc);
  }

  const int mAdd = (lane >> 4) << 3;
#pragma unroll
  for (int v = 0; v < 8; ++v) {
    int m = mBase + v + mAdd;
    out[(size_t)m * GH + col] = (__bf16)acc[v];
  }
}

// Persistent recurrent scan. One block per 16 batch rows (grid = B/16 = 4),
// 1024 threads = 32 waves; wave w owns output columns [16w, 16w+16).
#define LDB (GH + 8)        // bf16 row stride (1040 B, 16B aligned, rotates banks)
#define LDF (GH + 4)        // f32 row stride  (2064 B, rotates banks)
#define XST (3 * 16 * GH)   // stage buffer elems per timestep (3 gates x 16 x 512)

__global__ __launch_bounds__(1024) void gru_scan(
    const __bf16* __restrict__ xzb, const __bf16* __restrict__ xrb,
    const __bf16* __restrict__ xhb,
    const __bf16* __restrict__ Uzb, const __bf16* __restrict__ Urb,
    const __bf16* __restrict__ Uhb,
    const float* __restrict__ cz, const float* __restrict__ cr,
    const float* __restrict__ ch,
    float* __restrict__ hT) {
  __shared__ __bf16 h_bf[16 * LDB];     // h (bf16) ; reused as r*h mid-step
  __shared__ float  h_f[16 * LDF];      // h (f32 master copy)
  __shared__ __bf16 stage[2 * XST];     // double-buffered x-projection tiles

  const int tid   = threadIdx.x;
  const int lane  = tid & 31;
  const int wave  = tid >> 5;
  const int bBase = blockIdx.x * 16;
  const int nBase = wave * 16;
  const int col   = nBase + (lane & 15);
  const int mAdd  = (lane >> 4) << 3;

  for (int i = tid; i < 16 * LDB; i += 1024) h_bf[i] = (__bf16)0.0f;
  for (int i = tid; i < 16 * LDF; i += 1024) h_f[i] = 0.0f;
  const float czv = cz[col], crv = cr[col], chv = ch[col];

  // async-stage timestep t into buffer: 3*16*64 = 3072 16B chunks, 3 per thread
  auto stage_step = [&](__bf16* buf, int t) {
#pragma unroll
    for (int c = 0; c < 3; ++c) {
      int id  = tid + c * 1024;       // 0..3071
      int g   = id >> 10;             // gate 0..2
      int r   = id & 1023;            // chunk within gate
      int m   = r >> 6;               // row 0..15
      int off = (r & 63) * 8;         // bf16 offset within row (16B chunks)
      const __bf16* src = (g == 0 ? xzb : (g == 1 ? xrb : xhb));
      src += ((size_t)(bBase + m) * GS + t) * GH + off;
      async_b128(buf + g * (16 * GH) + m * GH + off, src);
    }
  };

  stage_step(stage, 0);
  wait_async0();
  __syncthreads();

  for (int t = 0; t < GS; ++t) {
    const __bf16* cur = stage + (t & 1) * XST;
    __bf16*       nxt = stage + ((t + 1) & 1) * XST;
    if (t + 1 < GS) stage_step(nxt, t + 1);  // overlap copy with the GEMMs

    // ---- z and r pre-activations: h @ Uz^T , h @ Ur^T ----
    v8f accZ = {};
    v8f accR = {};
#pragma unroll 4
    for (int k0 = 0; k0 < GH; k0 += 32) {
      v16bf a  = load_A_lds(h_bf, lane, k0, LDB);
      v16bf bz = load_B_w(Uzb, nBase, k0, GH, lane);
      v16bf br = load_B_w(Urb, nBase, k0, GH, lane);
      accZ = wmma_bf16(a, bz, accZ);
      accR = wmma_bf16(a, br, accR);
    }
    __syncthreads();  // all waves done reading h_bf before it becomes r*h

    float zv[8];
#pragma unroll
    for (int v = 0; v < 8; ++v) {
      int m = v + mAdd;
      float z = fast_sigmoid(accZ[v] + (float)cur[0 * 16 * GH + m * GH + col] + czv);
      float r = fast_sigmoid(accR[v] + (float)cur[1 * 16 * GH + m * GH + col] + crv);
      zv[v] = z;
      h_bf[m * LDB + col] = (__bf16)(r * h_f[m * LDF + col]);  // own columns
    }
    __syncthreads();  // r*h visible to all waves

    // ---- candidate: (r*h) @ Uh^T ----
    v8f accH = {};
#pragma unroll 4
    for (int k0 = 0; k0 < GH; k0 += 32) {
      v16bf a  = load_A_lds(h_bf, lane, k0, LDB);
      v16bf bh = load_B_w(Uhb, nBase, k0, GH, lane);
      accH = wmma_bf16(a, bh, accH);
    }
    __syncthreads();  // all waves done reading r*h before h_bf becomes h_new

#pragma unroll
    for (int v = 0; v < 8; ++v) {
      int m = v + mAdd;
      float htil = fast_tanh(accH[v] + (float)cur[2 * 16 * GH + m * GH + col] + chv);
      float hold = h_f[m * LDF + col];
      float hnew = (1.0f - zv[v]) * hold + zv[v] * htil;
      h_f[m * LDF + col]  = hnew;                // own columns
      h_bf[m * LDB + col] = (__bf16)hnew;        // own columns
    }
    wait_async0();    // this wave's stage copies for t+1 are in LDS
    __syncthreads();  // h_new + everyone's stage copies visible
  }

#pragma unroll
  for (int v = 0; v < 8; ++v) {
    int m = v + mAdd;
    hT[(size_t)(bBase + m) * GH + col] = h_f[m * LDF + col];
  }
}

// Final head: out = h_T @ Wf^T + bf  ([64x512] x [512x10])
__global__ void gru_head(const float* __restrict__ hT,
                         const float* __restrict__ Wf,
                         const float* __restrict__ bfv,
                         float* __restrict__ out) {
  int i = blockIdx.x * blockDim.x + threadIdx.x;
  if (i >= GB * GO) return;
  int b = i / GO, o = i % GO;
  float s = bfv[o];
  const float* hp = hT + (size_t)b * GH;
  const float* wp = Wf + (size_t)o * GH;
#pragma unroll 8
  for (int k = 0; k < GH; ++k) s += hp[k] * wp[k];
  out[i] = s;
}

// ---------- launch ----------

extern "C" void kernel_launch(void* const* d_in, const int* in_sizes, int n_in,
                              void* d_out, int out_size, void* d_ws,
                              size_t ws_size, hipStream_t stream) {
  (void)in_sizes; (void)n_in; (void)out_size; (void)ws_size;
  const float* x  = (const float*)d_in[0];
  const float* Wz = (const float*)d_in[1];
  const float* bz = (const float*)d_in[2];
  const float* Uz = (const float*)d_in[3];
  const float* cz = (const float*)d_in[4];
  const float* Wr = (const float*)d_in[5];
  const float* br = (const float*)d_in[6];
  const float* Ur = (const float*)d_in[7];
  const float* cr = (const float*)d_in[8];
  const float* Wh = (const float*)d_in[9];
  const float* bh = (const float*)d_in[10];
  const float* Uh = (const float*)d_in[11];
  const float* ch = (const float*)d_in[12];
  const float* Wf = (const float*)d_in[13];
  const float* bf = (const float*)d_in[14];
  float* out = (float*)d_out;

  // workspace carve-up
  char* p = (char*)d_ws;
  const size_t szU = (size_t)GH * GH * sizeof(__bf16);       // 512 KB
  const size_t szW = (size_t)GH * GI * sizeof(__bf16);       // 256 KB
  const size_t szX = (size_t)GB * GS * GH * sizeof(__bf16);  // 128 MB
  __bf16* Uzb = (__bf16*)p; p += szU;
  __bf16* Urb = (__bf16*)p; p += szU;
  __bf16* Uhb = (__bf16*)p; p += szU;
  __bf16* Wzb = (__bf16*)p; p += szW;
  __bf16* Wrb = (__bf16*)p; p += szW;
  __bf16* Whb = (__bf16*)p; p += szW;
  __bf16* xzb = (__bf16*)p; p += szX;
  __bf16* xrb = (__bf16*)p; p += szX;
  __bf16* xhb = (__bf16*)p; p += szX;
  float*  hT  = (float*)p;  p += (size_t)GB * GH * sizeof(float);

  // 1) convert weights to bf16
  {
    int nU = GH * GH, nW = GH * GI;
    dim3 blk(256);
    gru_cvt_bf16<<<dim3((nW + 255) / 256), blk, 0, stream>>>(Wz, Wzb, nW);
    gru_cvt_bf16<<<dim3((nW + 255) / 256), blk, 0, stream>>>(Wr, Wrb, nW);
    gru_cvt_bf16<<<dim3((nW + 255) / 256), blk, 0, stream>>>(Wh, Whb, nW);
    gru_cvt_bf16<<<dim3((nU + 255) / 256), blk, 0, stream>>>(Uz, Uzb, nU);
    gru_cvt_bf16<<<dim3((nU + 255) / 256), blk, 0, stream>>>(Ur, Urb, nU);
    gru_cvt_bf16<<<dim3((nU + 255) / 256), blk, 0, stream>>>(Uh, Uhb, nU);
  }

  // 2) input projections via WMMA: (B*S/16)*(H/16) tiles / 8 waves per block
  {
    int nTiles = (GB * GS / 16) * (GH / 16);  // 262144
    dim3 grid(nTiles / 8, 3);
    gru_proj<<<grid, 256, 0, stream>>>(x, Wzb, Wrb, Whb, bz, br, bh,
                                       xzb, xrb, xhb);
  }

  // 3) recurrent scan: 4 independent batch groups of 16 rows
  gru_scan<<<dim3(GB / 16), 1024, 0, stream>>>(xzb, xrb, xhb, Uzb, Urb, Uhb,
                                               cz, cr, ch, hT);

  // 4) output head
  gru_head<<<dim3((GB * GO + 255) / 256), 256, 0, stream>>>(hT, Wf, bf, out);
}
